// model_2_68453188764358
// MI455X (gfx1250) — matrix operand order repaired and checked by reading, not compiled, not run on hardware
//
#include <hip/hip_runtime.h>
#include <math.h>

// ---------------- types & helpers ----------------
typedef __attribute__((ext_vector_type(16))) __bf16 v16bf;
typedef __attribute__((ext_vector_type(8)))  float  v8f;

#define WMMA_BF16(a,b,c) \
  __builtin_amdgcn_wmma_f32_16x16x32_bf16(false,(a),false,(b),(short)0,(c),false,false)

union BFU { unsigned short u; __bf16 b; };
static __device__ inline __bf16 u2bf(unsigned short s){ BFU x; x.u = s; return x.b; }
static __device__ inline float sigf(float x){ return 1.f/(1.f+__expf(-x)); }

// model dims
#define BB   32
#define TT   512
#define JJ   64
#define WCH  16
#define DD   128
#define CEV  16
#define HID  128
#define GAT  512   // 4*HID

// ======================================================================
// Generic WMMA GEMM (templated: straight-line epilogue per instantiation)
//   C[b] = act( A[b] @ B[b]^T + biases )
//   A: (M,K) f32 row-major.  B: (N,K) f32 row-major (y = x @ W^T).
//   Block = 256 thr = 8 waves = 64x32 output tile.
//   Next-k-tile global_prefetch_b8 issued each iteration.
// ======================================================================
template<int ACT, int NBIAS>
__global__ __launch_bounds__(256)
void k_gemm(const float* __restrict__ A, int lda, long sA,
            const float* __restrict__ Bm, int ldb, long sB,
            const float* __restrict__ bias0, const float* __restrict__ bias1,
            float* __restrict__ C, int ldc, long sC,
            int M, int N, int K)
{
  int bz = blockIdx.z;
  A  += bz * sA;  Bm += bz * sB;  C += bz * sC;
  int tid = threadIdx.x, wid = tid >> 5, lane = tid & 31;
  int l16 = lane & 15, hh = lane >> 4;
  int row0 = blockIdx.y * 64 + (wid & 3) * 16;
  int col0 = blockIdx.x * 32 + (wid >> 2) * 16;
  if (row0 >= M || col0 >= N) return;

  v8f acc = {0.f,0.f,0.f,0.f,0.f,0.f,0.f,0.f};
  int am = row0 + l16;
  int bn = col0 + l16;
  const float* Arow = A  + (long)am * lda;
  const float* Brow = Bm + (long)bn * ldb;

  for (int kb = 0; kb < K; kb += 32) {
    const float* Ar = Arow + kb;
    const float* Br = Brow + kb + hh*16;
    if (kb + 32 < K) {                       // gfx1250 global_prefetch_b8
      __builtin_prefetch(Ar + 32, 0, 1);
      __builtin_prefetch(Br + 32, 0, 1);
    }
    v16bf af;
#pragma unroll
    for (int v = 0; v < 4; ++v) {
      int k0 = hh*16 + 2*v;
      float2 f0 = *(const float2*)(Ar + k0);
      float2 f1 = *(const float2*)(Ar + 8 + k0);
      af[2*v]     = (__bf16)f0.x;  af[2*v+1]   = (__bf16)f0.y;
      af[8+2*v]   = (__bf16)f1.x;  af[8+2*v+1] = (__bf16)f1.y;
    }
    v16bf bf;
#pragma unroll
    for (int v = 0; v < 8; ++v) {
      float2 f = *(const float2*)(Br + 2*v);
      bf[2*v] = (__bf16)f.x;  bf[2*v+1] = (__bf16)f.y;
    }
    acc = WMMA_BF16(af, bf, acc);
  }

  float bs = 0.f;
  if (NBIAS >= 1) bs += bias0[bn];
  if (NBIAS >= 2) bs += bias1[bn];
#pragma unroll
  for (int r = 0; r < 8; ++r) {
    int cm = row0 + r + 8*hh;
    float v = acc[r] + bs;
    if      (ACT == 1) v = sigf(v);
    else if (ACT == 2) v = fmaxf(v, 0.f);
    C[(long)cm * ldc + bn] = v;
  }
}

// ======================================================================
// Char-CNN + maxpool + word-embedding gather.  Block = 16 tokens.
// Width loop fully unrolled (kw/Kdim compile-time); boundary handling is
// clamp-address + select (no divergent loads, no exec-mask churn).
// ======================================================================
__global__ __launch_bounds__(256)
void k_embed(const int* __restrict__ chars, const int* __restrict__ words,
             const float* __restrict__ char_emb, const float* __restrict__ word_emb,
             const float* __restrict__ cw3, const float* __restrict__ cw5,
             const float* __restrict__ cw7,
             const float* __restrict__ cb3, const float* __restrict__ cb5,
             const float* __restrict__ cb7,
             float* __restrict__ act)
{
  __shared__ __bf16 emb[16 * WCH * CEV];   // [tok][char][ce]  8 KB
  int tokBase = blockIdx.x * 16;
  int tid = threadIdx.x;
  {
    int i = tid >> 4, j = tid & 15;
    int id = chars[(tokBase + i) * WCH + j];
    const float* e = char_emb + (long)id * CEV;
#pragma unroll
    for (int c = 0; c < CEV; ++c) emb[(i*WCH + j)*CEV + c] = (__bf16)e[c];
  }
  __syncthreads();

  int wid = tid >> 5, lane = tid & 31, l16 = lane & 15, hh = lane >> 4;
  float maxv[8];
#pragma unroll
  for (int r = 0; r < 8; ++r) maxv[r] = -1e30f;

  const float* cws[3] = {cw3, cw5, cw7};
  const float* cbs[3] = {cb3, cb5, cb7};
  const int    kws[3] = {3, 5, 7};

#pragma unroll
  for (int wi = 0; wi < 3; ++wi) {
    const int kw   = kws[wi];             // compile-time after unroll
    const int Kdim = CEV * kw;
    const int nk   = (Kdim + 31) >> 5;
    const int npos = WCH - kw + 1;
    int ch = wid*16 + l16;                // A-frag row = channel
    const float* W = cws[wi];

    // per-r bias, hoisted out of the position loop
    float cb[8];
#pragma unroll
    for (int r = 0; r < 8; ++r) cb[r] = cbs[wi][wid*16 + r + 8*hh];

    // A-fragments (weights): unconditional clamped loads + select
    v16bf af[4];
#pragma unroll
    for (int kc = 0; kc < 4; ++kc) {
      if (kc >= nk) break;
#pragma unroll
      for (int v = 0; v < 4; ++v) {
        int k0 = kc*32 + hh*16 + 2*v, k1 = k0 + 8;
        int k0c = (k0   < Kdim) ? k0   : 0;
        int k1c = (k1   < Kdim) ? k1   : 0;
        float w0 = W[(long)ch*Kdim + k0c];
        float w1 = W[(long)ch*Kdim + k0c + 1];
        float w2 = W[(long)ch*Kdim + k1c];
        float w3 = W[(long)ch*Kdim + ((k1+1 < Kdim) ? k1c + 1 : 0)];
        af[kc][2*v]     = (__bf16)((k0   < Kdim) ? w0 : 0.f);
        af[kc][2*v+1]   = (__bf16)((k0+1 < Kdim) ? w1 : 0.f);
        af[kc][8+2*v]   = (__bf16)((k1   < Kdim) ? w2 : 0.f);
        af[kc][8+2*v+1] = (__bf16)((k1+1 < Kdim) ? w3 : 0.f);
      }
    }

    for (int p = 0; p < npos; ++p) {
      v8f acc = {0.f,0.f,0.f,0.f,0.f,0.f,0.f,0.f};
      int tok = l16;                      // B-frag col = token
#pragma unroll
      for (int kc = 0; kc < 4; ++kc) {
        if (kc >= nk) break;
        v16bf bf;
#pragma unroll
        for (int v = 0; v < 8; ++v) {
          int k  = kc*32 + hh*16 + 2*v;
          int ka = (k   < Kdim) ? k   : 0;
          int kb = (k+1 < Kdim) ? k+1 : 0;
          int c0 = ka / kw, d0 = ka - c0*kw;    // kw const -> strength-reduced
          int c1 = kb / kw, d1 = kb - c1*kw;
          __bf16 e0 = emb[(tok*WCH + p + d0)*CEV + c0];
          __bf16 e1 = emb[(tok*WCH + p + d1)*CEV + c1];
          bf[2*v]   = (k   < Kdim) ? e0 : (__bf16)0.f;
          bf[2*v+1] = (k+1 < Kdim) ? e1 : (__bf16)0.f;
        }
        acc = WMMA_BF16(af[kc], bf, acc);
      }
#pragma unroll
      for (int r = 0; r < 8; ++r)
        maxv[r] = fmaxf(maxv[r], acc[r] + cb[r]);
    }
  }
#pragma unroll
  for (int r = 0; r < 8; ++r) {
    int chn = wid*16 + r + 8*hh;
    act[(long)(tokBase + l16) * 256 + chn] = maxv[r];
  }
  for (int u = tid; u < 16 * DD; u += 256) {
    int i = u >> 7, d = u & 127;
    int wd = words[tokBase + i];
    act[(long)(tokBase + i) * 256 + DD + d] = word_emb[(long)wd * DD + d];
  }
}

// ======================================================================
// Persistent BiLSTM recurrence. grid.x = 2 (fwd/bwd), block = 1024 (32 waves).
//  - Whh bf16 fragments resident in registers (2 tiles x 4 kc x 8 VGPRs)
//  - next-step xg staged into a single LDS buffer via async global->LDS b128
//    (ASYNCcnt / s_wait_asynccnt), overlapped with the cell update
// ======================================================================
__device__ inline void stage_xg_async(const float* __restrict__ xg, int te,
                                      int tid, float* __restrict__ xbuf, int Tn)
{
#pragma unroll
  for (int i = 0; i < 4; ++i) {
    int vu = tid + i * 1024;               // vec4 index in [0, 4096)
    int b  = vu >> 7;                      // batch row
    int n4 = vu & 127;                     // vec4 within gate row
    const float* ga = xg + ((long)b * Tn + te) * GAT + n4 * 4;
    unsigned lds   = (unsigned)(unsigned long long)(&xbuf[vu * 4]);
    unsigned long long g64 = (unsigned long long)ga;
    asm volatile("global_load_async_to_lds_b128 %0, %1, off"
                 :: "v"(lds), "v"(g64) : "memory");
  }
}

__global__ __launch_bounds__(1024)
void k_lstm(const float* __restrict__ xgf, const float* __restrict__ xgb,
            const float* __restrict__ Whhf, const float* __restrict__ Whhb,
            float* __restrict__ Hout, int Tn)
{
  int dir = blockIdx.x;
  const float* xg  = dir ? xgb  : xgf;
  const float* Whh = dir ? Whhb : Whhf;

  __shared__ __bf16 hbuf[BB * HID];                 //  8 KB
  __shared__ float  cbuf[BB * HID];                 // 16 KB
  __shared__ float  gbuf[BB * GAT];                 // 64 KB
  __shared__ __align__(16) float xbuf[BB * GAT];    // 64 KB async-staged xg
  int tid = threadIdx.x;
  for (int u = tid; u < BB*HID; u += 1024) { hbuf[u] = (__bf16)0.f; cbuf[u] = 0.f; }

  // stage xg for step 0 through the async global->LDS path
  stage_xg_async(xg, dir ? (Tn - 1) : 0, tid, xbuf, Tn);
  asm volatile("s_wait_asynccnt 0x0" ::: "memory");
  __syncthreads();

  int wid = tid >> 5, lane = tid & 31, l16 = lane & 15, hh = lane >> 4;
  const unsigned int* h32 = (const unsigned int*)hbuf;

  // this wave owns two gate tiles with a common batch tile
  int tile0 = wid * 2;
  int row0  = (tile0 >> 5) * 16;            // batch rows
  int colA  = (tile0 & 31) * 16;
  int colB  = ((tile0 + 1) & 31) * 16;
  int am = row0 + l16;
  int bnA = colA + l16, bnB = colB + l16;

  // ---- preload Whh fragments into registers (bf16), resident all steps ----
  v16bf bwA[4], bwB[4];
#pragma unroll
  for (int kc = 0; kc < 4; ++kc) {
    const float* BrA = Whh + (long)bnA * HID + kc*32 + hh*16;
    const float* BrB = Whh + (long)bnB * HID + kc*32 + hh*16;
#pragma unroll
    for (int v = 0; v < 8; ++v) {
      float2 fa = *(const float2*)(BrA + 2*v);
      float2 fb = *(const float2*)(BrB + 2*v);
      bwA[kc][2*v] = (__bf16)fa.x;  bwA[kc][2*v+1] = (__bf16)fa.y;
      bwB[kc][2*v] = (__bf16)fb.x;  bwB[kc][2*v+1] = (__bf16)fb.y;
    }
  }

  for (int t = 0; t < Tn; ++t) {
    int te = dir ? (Tn - 1 - t) : t;

    // A-fragments (h) from LDS — shared by both tiles
    v16bf af[4];
#pragma unroll
    for (int kc = 0; kc < 4; ++kc) {
#pragma unroll
      for (int v = 0; v < 4; ++v) {
        int k0 = kc*32 + hh*16 + 2*v;
        unsigned int p0 = h32[(am*HID + k0)      >> 1];
        unsigned int p1 = h32[(am*HID + k0 + 8) >> 1];
        af[kc][2*v]     = u2bf((unsigned short)(p0 & 0xffff));
        af[kc][2*v+1]   = u2bf((unsigned short)(p0 >> 16));
        af[kc][8+2*v]   = u2bf((unsigned short)(p1 & 0xffff));
        af[kc][8+2*v+1] = u2bf((unsigned short)(p1 >> 16));
      }
    }
    // tile A
    {
      v8f acc;
#pragma unroll
      for (int r = 0; r < 8; ++r)
        acc[r] = xbuf[(row0 + r + 8*hh) * GAT + bnA];
#pragma unroll
      for (int kc = 0; kc < 4; ++kc) acc = WMMA_BF16(af[kc], bwA[kc], acc);
#pragma unroll
      for (int r = 0; r < 8; ++r)
        gbuf[(row0 + r + 8*hh) * GAT + bnA] = acc[r];
    }
    // tile B
    {
      v8f acc;
#pragma unroll
      for (int r = 0; r < 8; ++r)
        acc[r] = xbuf[(row0 + r + 8*hh) * GAT + bnB];
#pragma unroll
      for (int kc = 0; kc < 4; ++kc) acc = WMMA_BF16(af[kc], bwB[kc], acc);
#pragma unroll
      for (int r = 0; r < 8; ++r)
        gbuf[(row0 + r + 8*hh) * GAT + bnB] = acc[r];
    }
    __syncthreads();   // xbuf fully consumed; gbuf complete

    // async-stage next step's xg while doing the cell update
    if (t + 1 < Tn)
      stage_xg_async(xg, dir ? (Tn - 2 - t) : (t + 1), tid, xbuf, Tn);

    for (int u = tid; u < BB*HID; u += 1024) {
      int b = u >> 7, d = u & 127;
      float gi = gbuf[b*GAT +           d];
      float gf = gbuf[b*GAT + HID     + d];
      float gg = gbuf[b*GAT + 2*HID   + d];
      float go = gbuf[b*GAT + 3*HID   + d];
      float c  = sigf(gf) * cbuf[u] + sigf(gi) * tanhf(gg);
      float h  = sigf(go) * tanhf(c);
      cbuf[u] = c;  hbuf[u] = (__bf16)h;
      Hout[((long)b * Tn + te) * (2*HID) + dir*HID + d] = h;
    }
    asm volatile("s_wait_asynccnt 0x0" ::: "memory");
    __syncthreads();
  }
}

// ---------------- small elementwise / reduction kernels ----------------
__global__ void k_hwcomb(float* __restrict__ x, const float* __restrict__ g,
                         const float* __restrict__ l, long n)
{
  long i = (long)blockIdx.x * blockDim.x + threadIdx.x;
  if (i < n) x[i] = g[i]*l[i] + (1.f - g[i]) * x[i];
}

__global__ void k_scale(const float* __restrict__ H, const float* __restrict__ w,
                        float* __restrict__ o, long n)
{
  long i = (long)blockIdx.x * blockDim.x + threadIdx.x;
  if (i < n) o[i] = H[i] * w[i & 255];
}

// transpose U (B,J,256) -> Ut (B,256,J) so attention GEMMs use the NK path
__global__ void k_transU(const float* __restrict__ U, float* __restrict__ Ut)
{
  long i = (long)blockIdx.x * blockDim.x + threadIdx.x;   // over B*J*256
  if (i >= (long)BB*JJ*256) return;
  int d = (int)(i & 255);
  long rj = i >> 8;
  int j = (int)(rj & (JJ-1));
  int b = (int)(rj >> 6);
  Ut[((long)b*256 + d)*JJ + j] = U[i];
}

__global__ void k_rowdot(const float* __restrict__ X, const float* __restrict__ w,
                         const float* __restrict__ b, float* __restrict__ y,
                         int K)
{
  int wid = threadIdx.x >> 5, lane = threadIdx.x & 31;
  int row = blockIdx.x * 8 + wid;
  float acc = 0.f;
  for (int k = lane; k < K; k += 32) acc += X[(long)row*K + k] * w[k];
#pragma unroll
  for (int off = 16; off; off >>= 1) acc += __shfl_xor(acc, off);
  if (lane == 0) y[row] = acc + b[0];
}

__global__ void k_softmax_j(float* __restrict__ S, const float* __restrict__ s1,
                            const float* __restrict__ s2, const float* __restrict__ c3,
                            float* __restrict__ mbuf)
{
  __shared__ float sh[JJ];
  int bt = blockIdx.x, b = bt >> 9, j = threadIdx.x;
  float v = S[(long)bt*JJ + j] + s1[bt] + s2[b*JJ + j] + c3[0];
  sh[j] = v; __syncthreads();
  for (int s = 32; s; s >>= 1) { if (j < s) sh[j] = fmaxf(sh[j], sh[j+s]); __syncthreads(); }
  float mx = sh[0]; __syncthreads();
  float e = __expf(v - mx);
  sh[j] = e; __syncthreads();
  for (int s = 32; s; s >>= 1) { if (j < s) sh[j] += sh[j+s]; __syncthreads(); }
  S[(long)bt*JJ + j] = e / sh[0];
  if (j == 0) mbuf[bt] = mx;
}

__global__ void k_softmax_t_ht(const float* __restrict__ mbuf,
                               const float* __restrict__ H, float* __restrict__ Ht)
{
  __shared__ float sh[TT];
  __shared__ float bbs[TT];
  int b = blockIdx.x, t = threadIdx.x;
  float v = mbuf[b*TT + t];
  sh[t] = v; __syncthreads();
  for (int s = 256; s; s >>= 1) { if (t < s) sh[t] = fmaxf(sh[t], sh[t+s]); __syncthreads(); }
  float mx = sh[0]; __syncthreads();
  float e = __expf(v - mx);
  sh[t] = e; bbs[t] = e; __syncthreads();
  for (int s = 256; s; s >>= 1) { if (t < s) sh[t] += sh[t+s]; __syncthreads(); }
  float inv = 1.f / sh[0]; __syncthreads();
  bbs[t] = bbs[t] * inv; __syncthreads();
  if (t < 256) {
    float acc = 0.f;
    for (int tt = 0; tt < TT; ++tt)
      acc += bbs[tt] * H[((long)b*TT + tt)*256 + t];
    Ht[b*256 + t] = acc;
  }
}

__global__ void k_buildG(const float* __restrict__ H, const float* __restrict__ Ua,
                         const float* __restrict__ Ht, float* __restrict__ G)
{
  long i = (long)blockIdx.x * blockDim.x + threadIdx.x;   // over B*T*256
  if (i >= (long)BB*TT*256) return;
  long row = i >> 8;  int d = (int)(i & 255);  int b = (int)(row >> 9);
  float h = H[i], u = Ua[i];
  float* g = G + row * 1024;
  g[d] = h;  g[256 + d] = u;  g[512 + d] = h*u;  g[768 + d] = h * Ht[b*256 + d];
}

__global__ void k_pout(const float* __restrict__ G, const float* __restrict__ M,
                       const float* __restrict__ pw, const float* __restrict__ pb,
                       float* __restrict__ out)
{
  int wid = threadIdx.x >> 5, lane = threadIdx.x & 31;
  int row = blockIdx.x * 8 + wid;
  float acc = 0.f;
  for (int k = lane; k < 1024; k += 32) acc += G[(long)row*1024 + k] * pw[k];
  for (int k = lane; k < 256;  k += 32) acc += M[(long)row*256  + k] * pw[1024 + k];
#pragma unroll
  for (int off = 16; off; off >>= 1) acc += __shfl_xor(acc, off);
  if (lane == 0) out[row] = acc + pb[0];
}

// ======================================================================
// orchestration
// ======================================================================
enum {
  I_CCHAR=0, I_QCHAR, I_CWORD, I_QWORD, I_CHAREMB, I_WORDEMB,
  I_CW3, I_CW5, I_CW7, I_CB3, I_CB5, I_CB7,
  I_HWGW, I_HWGB, I_HWLW, I_HWLB,
  I_CTXF_WIH, I_CTXF_WHH, I_CTXF_BIH, I_CTXF_BHH,
  I_CTXB_WIH, I_CTXB_WHH, I_CTXB_BIH, I_CTXB_BHH,
  I_SIM1W, I_SIM1B, I_SIM2W, I_SIM2B, I_SIM3W, I_SIM3B,
  I_M0F_WIH, I_M0F_WHH, I_M0F_BIH, I_M0F_BHH,
  I_M0B_WIH, I_M0B_WHH, I_M0B_BIH, I_M0B_BHH,
  I_M1F_WIH, I_M1F_WHH, I_M1F_BIH, I_M1F_BHH,
  I_M1B_WIH, I_M1B_WHH, I_M1B_BIH, I_M1B_BHH,
  I_P1W, I_P1B, I_P2W, I_P2B,
  I_OF_WIH, I_OF_WHH, I_OF_BIH, I_OF_BHH,
  I_OB_WIH, I_OB_WHH, I_OB_BIH, I_OB_BHH
};

extern "C" void kernel_launch(void* const* d_in, const int* in_sizes, int n_in,
                              void* d_out, int out_size, void* d_ws, size_t ws_size,
                              hipStream_t stream)
{
  (void)in_sizes; (void)n_in; (void)out_size; (void)ws_size;
  auto F = [&](int i){ return (const float*)d_in[i]; };
  auto I = [&](int i){ return (const int*)d_in[i]; };

  const long RC = (long)BB*TT;      // 16384 context rows
  const long RQ = (long)BB*JJ;      // 2048 question rows

  float* ws = (float*)d_ws;
  float* ACTC = ws;                 ws += RC*256;
  float* ACTQ = ws;                 ws += RQ*256;
  float* TMP1 = ws;                 ws += RC*256;   // gate / Hs / Ua
  float* TMP2 = ws;                 ws += RC*256;   // lin / M1mid / M2
  float* XGF  = ws;                 ws += RC*GAT;
  float* XGB  = ws;                 ws += RC*GAT;
  float* Hc   = ws;                 ws += RC*256;
  float* Uq   = ws;                 ws += RQ*256;
  float* Ut   = ws;                 ws += RQ*256;   // U transposed (B,256,J)
  float* Sm   = ws;                 ws += RC*JJ;
  float* Gm   = ws;                 ws += RC*1024;
  float* M1   = ws;                 ws += RC*256;
  float* S1   = ws;                 ws += RC;
  float* S2   = ws;                 ws += RQ;
  float* Mb   = ws;                 ws += RC;
  float* Ht   = ws;                 ws += BB*256;

  float* OUT1 = (float*)d_out;
  float* OUT2 = OUT1 + RC;

  dim3 b256(256), b1024(1024);

  // 1) embeddings (char-CNN WMMA + word gather)
  k_embed<<<dim3(RC/16), b256, 0, stream>>>(I(I_CCHAR), I(I_CWORD), F(I_CHAREMB),
      F(I_WORDEMB), F(I_CW3), F(I_CW5), F(I_CW7), F(I_CB3), F(I_CB5), F(I_CB7), ACTC);
  k_embed<<<dim3(RQ/16), b256, 0, stream>>>(I(I_QCHAR), I(I_QWORD), F(I_CHAREMB),
      F(I_WORDEMB), F(I_CW3), F(I_CW5), F(I_CW7), F(I_CB3), F(I_CB5), F(I_CB7), ACTQ);

  // 2) highway (2 layers, context + question)
  for (int l = 0; l < 2; ++l) {
    const float* gw = F(I_HWGW) + (long)l*256*256;
    const float* gb = F(I_HWGB) + l*256;
    const float* lw = F(I_HWLW) + (long)l*256*256;
    const float* lb = F(I_HWLB) + l*256;
    k_gemm<1,1><<<dim3(256/32, RC/64, 1), b256, 0, stream>>>(ACTC,256,0, gw,256,0,
        gb,nullptr, TMP1,256,0, (int)RC,256,256);
    k_gemm<2,1><<<dim3(256/32, RC/64, 1), b256, 0, stream>>>(ACTC,256,0, lw,256,0,
        lb,nullptr, TMP2,256,0, (int)RC,256,256);
    k_hwcomb<<<dim3((RC*256+255)/256), b256, 0, stream>>>(ACTC, TMP1, TMP2, RC*256);
    k_gemm<1,1><<<dim3(256/32, RQ/64, 1), b256, 0, stream>>>(ACTQ,256,0, gw,256,0,
        gb,nullptr, TMP1,256,0, (int)RQ,256,256);
    k_gemm<2,1><<<dim3(256/32, RQ/64, 1), b256, 0, stream>>>(ACTQ,256,0, lw,256,0,
        lb,nullptr, TMP2,256,0, (int)RQ,256,256);
    k_hwcomb<<<dim3((RQ*256+255)/256), b256, 0, stream>>>(ACTQ, TMP1, TMP2, RQ*256);
  }

  // 3) contextual BiLSTM: H = bilstm(ACTC), U = bilstm(ACTQ)
  k_gemm<0,2><<<dim3(GAT/32, RC/64, 1), b256, 0, stream>>>(ACTC,256,0, F(I_CTXF_WIH),256,0,
      F(I_CTXF_BIH), F(I_CTXF_BHH), XGF,GAT,0, (int)RC,GAT,256);
  k_gemm<0,2><<<dim3(GAT/32, RC/64, 1), b256, 0, stream>>>(ACTC,256,0, F(I_CTXB_WIH),256,0,
      F(I_CTXB_BIH), F(I_CTXB_BHH), XGB,GAT,0, (int)RC,GAT,256);
  k_lstm<<<dim3(2), b1024, 0, stream>>>(XGF, XGB, F(I_CTXF_WHH), F(I_CTXB_WHH), Hc, TT);

  k_gemm<0,2><<<dim3(GAT/32, RQ/64, 1), b256, 0, stream>>>(ACTQ,256,0, F(I_CTXF_WIH),256,0,
      F(I_CTXF_BIH), F(I_CTXF_BHH), XGF,GAT,0, (int)RQ,GAT,256);
  k_gemm<0,2><<<dim3(GAT/32, RQ/64, 1), b256, 0, stream>>>(ACTQ,256,0, F(I_CTXB_WIH),256,0,
      F(I_CTXB_BIH), F(I_CTXB_BHH), XGB,GAT,0, (int)RQ,GAT,256);
  k_lstm<<<dim3(2), b1024, 0, stream>>>(XGF, XGB, F(I_CTXF_WHH), F(I_CTXB_WHH), Uq, JJ);

  // 4) similarity + attention
  k_rowdot<<<dim3(RC/8), b256, 0, stream>>>(Hc, F(I_SIM1W), F(I_SIM1B), S1, 256);
  k_rowdot<<<dim3(RQ/8), b256, 0, stream>>>(Uq, F(I_SIM2W), F(I_SIM2B), S2, 256);
  k_scale<<<dim3((RC*256+255)/256), b256, 0, stream>>>(Hc, F(I_SIM3W), TMP1, RC*256);
  k_transU<<<dim3((RQ*256+255)/256), b256, 0, stream>>>(Uq, Ut);
  // S_core[b] = Hs[b] (512x256) @ U[b]^T   (U as (J,256) NK layout)
  k_gemm<0,0><<<dim3(JJ/32, TT/64, BB), b256, 0, stream>>>(TMP1,256,(long)TT*256,
      Uq,256,(long)JJ*256, nullptr,nullptr, Sm,JJ,(long)TT*JJ, TT,JJ,256);
  k_softmax_j<<<dim3(RC), dim3(JJ), 0, stream>>>(Sm, S1, S2, F(I_SIM3B), Mb);
  k_softmax_t_ht<<<dim3(BB), dim3(TT), 0, stream>>>(Mb, Hc, Ht);
  // Ua[b] = a[b] (512x64) @ Ut[b]^T   (Ut as (256,J) NK layout)
  k_gemm<0,0><<<dim3(256/32, TT/64, BB), b256, 0, stream>>>(Sm,JJ,(long)TT*JJ,
      Ut,JJ,(long)256*JJ, nullptr,nullptr, TMP1,256,(long)TT*256, TT,256,JJ);
  k_buildG<<<dim3((RC*256+255)/256), b256, 0, stream>>>(Hc, TMP1, Ht, Gm);

  // 5) modeling layers
  k_gemm<0,2><<<dim3(GAT/32, RC/64, 1), b256, 0, stream>>>(Gm,1024,0, F(I_M0F_WIH),1024,0,
      F(I_M0F_BIH), F(I_M0F_BHH), XGF,GAT,0, (int)RC,GAT,1024);
  k_gemm<0,2><<<dim3(GAT/32, RC/64, 1), b256, 0, stream>>>(Gm,1024,0, F(I_M0B_WIH),1024,0,
      F(I_M0B_BIH), F(I_M0B_BHH), XGB,GAT,0, (int)RC,GAT,1024);
  k_lstm<<<dim3(2), b1024, 0, stream>>>(XGF, XGB, F(I_M0F_WHH), F(I_M0B_WHH), TMP2, TT);

  k_gemm<0,2><<<dim3(GAT/32, RC/64, 1), b256, 0, stream>>>(TMP2,256,0, F(I_M1F_WIH),256,0,
      F(I_M1F_BIH), F(I_M1F_BHH), XGF,GAT,0, (int)RC,GAT,256);
  k_gemm<0,2><<<dim3(GAT/32, RC/64, 1), b256, 0, stream>>>(TMP2,256,0, F(I_M1B_WIH),256,0,
      F(I_M1B_BIH), F(I_M1B_BHH), XGB,GAT,0, (int)RC,GAT,256);
  k_lstm<<<dim3(2), b1024, 0, stream>>>(XGF, XGB, F(I_M1F_WHH), F(I_M1B_WHH), M1, TT);

  // 6) p1
  k_pout<<<dim3(RC/8), b256, 0, stream>>>(Gm, M1, F(I_P1W), F(I_P1B), OUT1);

  // 7) output LSTM + p2
  k_gemm<0,2><<<dim3(GAT/32, RC/64, 1), b256, 0, stream>>>(M1,256,0, F(I_OF_WIH),256,0,
      F(I_OF_BIH), F(I_OF_BHH), XGF,GAT,0, (int)RC,GAT,256);
  k_gemm<0,2><<<dim3(GAT/32, RC/64, 1), b256, 0, stream>>>(M1,256,0, F(I_OB_WIH),256,0,
      F(I_OB_BIH), F(I_OB_BHH), XGB,GAT,0, (int)RC,GAT,256);
  k_lstm<<<dim3(2), b1024, 0, stream>>>(XGF, XGB, F(I_OF_WHH), F(I_OB_WHH), TMP2, TT);

  k_pout<<<dim3(RC/8), b256, 0, stream>>>(Gm, TMP2, F(I_P2W), F(I_P2B), OUT2);
}
